// AGGemmIntraNode_31997506355744
// MI455X (gfx1250) — matrix operand order repaired and checked by reading, not compiled, not run on hardware
//
#include <hip/hip_runtime.h>

// ---------------------------------------------------------------------------
// AG-GEMM for MI455X (gfx1250, wave32):
//   C[8192,4096] = A[8192,4096] @ W[4096,4096]^T   (all f32)
// All-gather is a no-op (shards contiguous). f32 emulated via bf16 hi/lo
// split: C += Ah*Wh + Al*Wh + Ah*Wl (3x V_WMMA_F32_16X16X32_BF16 per k-step).
//
// Fast path (needs 201 MB of d_ws):
//   pass 1: bandwidth-bound split  f32 -> (bf16 hi, bf16 lo) into d_ws
//   pass 2: GEMM, workgroup tile 128x256, 3-stage async pipeline:
//           global_load_async_to_lds_b128 (ASYNCcnt) -> ds_load_b128 -> WMMA
//           48 WMMAs per 32 ds_load_b128 per wave per k-step.
// Fallback (small d_ws): single-kernel inline-split version.
// ---------------------------------------------------------------------------

typedef __attribute__((ext_vector_type(16))) __bf16 v16bf;
typedef __attribute__((ext_vector_type(8)))  __bf16 v8bf;
typedef __attribute__((ext_vector_type(8)))  float  v8f;
typedef __attribute__((ext_vector_type(4)))  float  v4f;

#define TILE_K 32
// fast-path GEMM geometry
#define GBM 128
#define GBN 256
// LDS element offsets within one stage (in __bf16 elements)
#define SA_HI 0
#define SA_LO (GBM * TILE_K)              // 4096
#define SW_HI (2 * GBM * TILE_K)          // 8192
#define SW_LO (SW_HI + GBN * TILE_K)      // 16384
#define STAGE_E (SW_LO + GBN * TILE_K)    // 24576 elements = 49152 bytes
#define STAGE_B (STAGE_E * 2)             // 49152 bytes
#define NSTAGES 3

__device__ __forceinline__ unsigned short f32_to_bf16_rne(float x) {
    unsigned int u = __builtin_bit_cast(unsigned int, x);
    unsigned int r = u + 0x7FFFu + ((u >> 16) & 1u);   // round-to-nearest-even
    return (unsigned short)(r >> 16);
}
__device__ __forceinline__ float bf16_bits_to_f32(unsigned short h) {
    unsigned int u = ((unsigned int)h) << 16;
    return __builtin_bit_cast(float, u);
}

// A-fragment (16x32 bf16, M x K) per CDNA5 ISA layout:
//   lanes 0-15 : row M=lane,    K = {0..7, 16..23}
//   lanes 16-31: row M=lane-16, K = {8..15, 24..31}
__device__ __forceinline__ v16bf load_fragA(const __bf16* base, int lane) {
    const int row  = lane & 15;
    const int koff = (lane >> 4) << 3;            // 0 or 8
    const __bf16* p = base + row * TILE_K + koff;
    v8bf c0 = *(const v8bf*)(p);
    v8bf c1 = *(const v8bf*)(p + 16);
    return __builtin_shufflevector(c0, c1, 0,1,2,3,4,5,6,7,8,9,10,11,12,13,14,15);
}

// B-fragment (32x16 bf16, K x N), column-major per lane (== rows of W[N,K]):
//   lanes 0-15 : col N=lane,    K = {0..7, 16..23};  lanes 16-31: col N=lane-16, K = {8..15, 24..31}
__device__ __forceinline__ v16bf load_fragB(const __bf16* base, int lane) {
    const int col  = lane & 15;
    const int koff = (lane >> 4) << 3;            // 0 or 8
    const __bf16* p = base + col * TILE_K + koff;
    v8bf c0 = *(const v8bf*)(p);
    v8bf c1 = *(const v8bf*)(p + 16);
    return __builtin_shufflevector(c0, c1, 0,1,2,3,4,5,6,7,8,9,10,11,12,13,14,15);
}

__device__ __forceinline__ v8f wmma_bf16(v16bf a, v16bf b, v8f c) {
    return __builtin_amdgcn_wmma_f32_16x16x32_bf16(
        false, a, false, b, (short)0, c, false, false);
}

// ---------------------------------------------------------------------------
// Pass 1: split f32 -> bf16 hi/lo. Each thread handles 8 contiguous elements.
// ---------------------------------------------------------------------------
__global__ void __launch_bounds__(256)
split_f32_bf16(const float* __restrict__ src, __bf16* __restrict__ hi,
               __bf16* __restrict__ lo) {
    const size_t i = ((size_t)blockIdx.x * 256 + threadIdx.x) * 8;
    v4f a = *(const v4f*)(src + i);
    v4f b = *(const v4f*)(src + i + 4);
    unsigned h[8], l[8];
#pragma unroll
    for (int e = 0; e < 4; ++e) {
        h[e]     = f32_to_bf16_rne(a[e]);
        l[e]     = f32_to_bf16_rne(a[e] - bf16_bits_to_f32((unsigned short)h[e]));
        h[e + 4] = f32_to_bf16_rne(b[e]);
        l[e + 4] = f32_to_bf16_rne(b[e] - bf16_bits_to_f32((unsigned short)h[e + 4]));
    }
    uint4 H, L;
    H.x = h[0] | (h[1] << 16);  H.y = h[2] | (h[3] << 16);
    H.z = h[4] | (h[5] << 16);  H.w = h[6] | (h[7] << 16);
    L.x = l[0] | (l[1] << 16);  L.y = l[2] | (l[3] << 16);
    L.z = l[4] | (l[5] << 16);  L.w = l[6] | (l[7] << 16);
    *(uint4*)(hi + i) = H;
    *(uint4*)(lo + i) = L;
}

// ---------------------------------------------------------------------------
// Pass 2: GEMM on pre-split bf16. Workgroup 128x256, 8 waves (2x4), each wave
// computes a 64x64 block of C. 3-stage async LDS pipeline (144 KB dynamic).
// ---------------------------------------------------------------------------
__global__ void __launch_bounds__(256)
ag_gemm_bf16_async(const __bf16* __restrict__ Ah, const __bf16* __restrict__ Al,
                   const __bf16* __restrict__ Wh, const __bf16* __restrict__ Wl,
                   float* __restrict__ C, int M, int N, int K) {
    extern __shared__ __align__(16) __bf16 lds[];   // NSTAGES * STAGE_E

    const int tid  = threadIdx.x;
    const int lane = tid & 31;
    const int wid  = tid >> 5;       // 0..7
    const int wr   = wid >> 2;       // 0..1  -> 64-row block
    const int wc   = wid & 3;        // 0..3  -> 64-col block

    const int m0 = blockIdx.y * GBM;
    const int n0 = blockIdx.x * GBN;

    v8f acc[4][4];
#pragma unroll
    for (int i = 0; i < 4; ++i)
#pragma unroll
        for (int j = 0; j < 4; ++j)
            acc[i][j] = (v8f)(0.0f);

    // --- async staging maps -------------------------------------------------
    // A tiles (128 rows x 64B): 2 threads/row, 2 x 16B chunks per thread.
    const int rA = tid >> 1;         // 0..127
    const int qA = tid & 1;          // chunk pair: k = qA*16 + {0,8}
    // W tiles (256 rows x 64B): 1 thread/row, 4 x 16B chunks per thread.
    const int rW = tid;              // 0..255

    const __bf16* pAh = Ah + (size_t)(m0 + rA) * K + qA * 16;
    const __bf16* pAl = Al + (size_t)(m0 + rA) * K + qA * 16;
    const __bf16* pWh = Wh + (size_t)(n0 + rW) * K;
    const __bf16* pWl = Wl + (size_t)(n0 + rW) * K;

    const unsigned ldsb  = (unsigned)(unsigned long long)&lds[0];
    const unsigned sbA   = ldsb + (unsigned)((rA * TILE_K + qA * 16) * 2);
    const unsigned sbW   = ldsb + (unsigned)(SW_HI * 2 + rW * TILE_K * 2);

    auto issue_stage = [&](int stage) {
        const unsigned a = sbA + (unsigned)stage * STAGE_B;
        const unsigned w = sbW + (unsigned)stage * STAGE_B;
        asm volatile("global_load_async_to_lds_b128 %0, %1, off"
                     :: "v"(a), "v"(pAh) : "memory");
        asm volatile("global_load_async_to_lds_b128 %0, %1, off"
                     :: "v"(a + 16u), "v"(pAh + 8) : "memory");
        asm volatile("global_load_async_to_lds_b128 %0, %1, off"
                     :: "v"(a + (unsigned)(SA_LO * 2)), "v"(pAl) : "memory");
        asm volatile("global_load_async_to_lds_b128 %0, %1, off"
                     :: "v"(a + (unsigned)(SA_LO * 2) + 16u), "v"(pAl + 8) : "memory");
        asm volatile("global_load_async_to_lds_b128 %0, %1, off"
                     :: "v"(w), "v"(pWh) : "memory");
        asm volatile("global_load_async_to_lds_b128 %0, %1, off"
                     :: "v"(w + 16u), "v"(pWh + 8) : "memory");
        asm volatile("global_load_async_to_lds_b128 %0, %1, off"
                     :: "v"(w + 32u), "v"(pWh + 16) : "memory");
        asm volatile("global_load_async_to_lds_b128 %0, %1, off"
                     :: "v"(w + 48u), "v"(pWh + 24) : "memory");
        asm volatile("global_load_async_to_lds_b128 %0, %1, off"
                     :: "v"(w + (unsigned)((SW_LO - SW_HI) * 2)), "v"(pWl) : "memory");
        asm volatile("global_load_async_to_lds_b128 %0, %1, off"
                     :: "v"(w + (unsigned)((SW_LO - SW_HI) * 2) + 16u), "v"(pWl + 8) : "memory");
        asm volatile("global_load_async_to_lds_b128 %0, %1, off"
                     :: "v"(w + (unsigned)((SW_LO - SW_HI) * 2) + 32u), "v"(pWl + 16) : "memory");
        asm volatile("global_load_async_to_lds_b128 %0, %1, off"
                     :: "v"(w + (unsigned)((SW_LO - SW_HI) * 2) + 48u), "v"(pWl + 24) : "memory");
        pAh += TILE_K; pAl += TILE_K; pWh += TILE_K; pWl += TILE_K;
    };

    const int nk = K / TILE_K;                 // 128
    issue_stage(0);                            // prefetch depth 2
    if (nk > 1) issue_stage(1);

    int cur = 0;
    for (int kt = 0; kt < nk; ++kt) {
        // prefetch stage kt+2 (overwrites the stage read at kt-1; safe after
        // the trailing barrier of iteration kt-1)
        if (kt + 2 < nk) {
            int nxt2 = cur + 2; if (nxt2 >= NSTAGES) nxt2 -= NSTAGES;
            issue_stage(nxt2);
        }
        // async loads complete in order; 12 issues per stage per thread:
        // waiting down to the N newest guarantees stage `cur` is in LDS.
        if (kt + 2 < nk)      asm volatile("s_wait_asynccnt 0x18" ::: "memory");
        else if (kt + 1 < nk) asm volatile("s_wait_asynccnt 0xC"  ::: "memory");
        else                  asm volatile("s_wait_asynccnt 0x0"  ::: "memory");
        __syncthreads();                       // all waves' stage landed

        const __bf16* S   = lds + cur * STAGE_E;
        const __bf16* LAh = S + SA_HI;
        const __bf16* LAl = S + SA_LO;
        const __bf16* LWh = S + SW_HI;
        const __bf16* LWl = S + SW_LO;

        // hold all A fragments (hi+lo) for this wave's 64 rows
        v16bf ah[4], al[4];
#pragma unroll
        for (int i = 0; i < 4; ++i) {
            const int mrow = (wr * 64 + i * 16) * TILE_K;
            ah[i] = load_fragA(LAh + mrow, lane);
            al[i] = load_fragA(LAl + mrow, lane);
        }
#pragma unroll
        for (int j = 0; j < 4; ++j) {
            const int nrow = (wc * 64 + j * 16) * TILE_K;
            v16bf bh = load_fragB(LWh + nrow, lane);
            v16bf bl = load_fragB(LWl + nrow, lane);
#pragma unroll
            for (int i = 0; i < 4; ++i) {
                acc[i][j] = wmma_bf16(ah[i], bh, acc[i][j]);   // hi*hi
                acc[i][j] = wmma_bf16(al[i], bh, acc[i][j]);   // lo*hi
                acc[i][j] = wmma_bf16(ah[i], bl, acc[i][j]);   // hi*lo
            }
        }
        __syncthreads();      // all waves done reading `cur` before overwrite
        ++cur; if (cur == NSTAGES) cur = 0;
    }

    // epilogue: C/D layout -> VGPR v, lanes 0-15: M=v, lanes 16-31: M=v+8
    const int lm = (lane >> 4) << 3;
    const int ln = lane & 15;
#pragma unroll
    for (int i = 0; i < 4; ++i)
#pragma unroll
        for (int j = 0; j < 4; ++j) {
            const int mm = m0 + wr * 64 + i * 16 + lm;
            const int nn = n0 + wc * 64 + j * 16 + ln;
#pragma unroll
            for (int v = 0; v < 8; ++v)
                C[(size_t)(mm + v) * N + nn] = acc[i][j][v];
        }
}

// ---------------------------------------------------------------------------
// Fallback: single-kernel version with inline hi/lo split (128x128 tile).
// ---------------------------------------------------------------------------
#define FBM 128
#define FBN 128

__global__ void __launch_bounds__(256)
ag_gemm_f32split_wmma(const float* __restrict__ A, const float* __restrict__ W,
                      float* __restrict__ C, int M, int N, int K) {
    __shared__ __align__(16) __bf16 sAhi[FBM * TILE_K];
    __shared__ __align__(16) __bf16 sAlo[FBM * TILE_K];
    __shared__ __align__(16) __bf16 sWhi[FBN * TILE_K];
    __shared__ __align__(16) __bf16 sWlo[FBN * TILE_K];

    const int tid  = threadIdx.x;
    const int lane = tid & 31;
    const int wid  = tid >> 5;
    const int wr   = wid >> 2;
    const int wc   = wid & 3;
    const int m0 = blockIdx.y * FBM;
    const int n0 = blockIdx.x * FBN;

    v8f acc[4][2];
#pragma unroll
    for (int i = 0; i < 4; ++i)
#pragma unroll
        for (int j = 0; j < 2; ++j)
            acc[i][j] = (v8f)(0.0f);

    const int lr = tid >> 1;
    const int lh = (tid & 1) << 4;

    for (int k0 = 0; k0 < K; k0 += TILE_K) {
        const float* ga = A + (size_t)(m0 + lr) * K + k0 + lh;
        const float* gw = W + (size_t)(n0 + lr) * K + k0 + lh;
#pragma unroll
        for (int i = 0; i < 16; i += 4) {
            v4f va = *(const v4f*)(ga + i);
            v4f vw = *(const v4f*)(gw + i);
            unsigned long long pAh = 0, pAl = 0, pWh = 0, pWl = 0;
#pragma unroll
            for (int e = 0; e < 4; ++e) {
                unsigned short ah = f32_to_bf16_rne(va[e]);
                unsigned short al = f32_to_bf16_rne(va[e] - bf16_bits_to_f32(ah));
                unsigned short wh = f32_to_bf16_rne(vw[e]);
                unsigned short wl = f32_to_bf16_rne(vw[e] - bf16_bits_to_f32(wh));
                pAh |= (unsigned long long)ah << (16 * e);
                pAl |= (unsigned long long)al << (16 * e);
                pWh |= (unsigned long long)wh << (16 * e);
                pWl |= (unsigned long long)wl << (16 * e);
            }
            const int c0 = lr * TILE_K + lh + i;
            *(unsigned long long*)(sAhi + c0) = pAh;
            *(unsigned long long*)(sAlo + c0) = pAl;
            *(unsigned long long*)(sWhi + c0) = pWh;
            *(unsigned long long*)(sWlo + c0) = pWl;
        }
        __syncthreads();

        v16bf bh[2], bl[2];
#pragma unroll
        for (int j = 0; j < 2; ++j) {
            const int nrow = (wc * 32 + j * 16) * TILE_K;
            bh[j] = load_fragB(sWhi + nrow, lane);
            bl[j] = load_fragB(sWlo + nrow, lane);
        }
#pragma unroll
        for (int i = 0; i < 4; ++i) {
            const int mrow = (wr * 64 + i * 16) * TILE_K;
            v16bf ah = load_fragA(sAhi + mrow, lane);
            v16bf al = load_fragA(sAlo + mrow, lane);
#pragma unroll
            for (int j = 0; j < 2; ++j) {
                acc[i][j] = wmma_bf16(ah, bh[j], acc[i][j]);
                acc[i][j] = wmma_bf16(al, bh[j], acc[i][j]);
                acc[i][j] = wmma_bf16(ah, bl[j], acc[i][j]);
            }
        }
        __syncthreads();
    }

    const int lm = (lane >> 4) << 3;
    const int ln = lane & 15;
#pragma unroll
    for (int i = 0; i < 4; ++i)
#pragma unroll
        for (int j = 0; j < 2; ++j) {
            const int mm = m0 + wr * 64 + i * 16 + lm;
            const int nn = n0 + wc * 32 + j * 16 + ln;
#pragma unroll
            for (int v = 0; v < 8; ++v)
                C[(size_t)(mm + v) * N + nn] = acc[i][j][v];
        }
}

extern "C" void kernel_launch(void* const* d_in, const int* in_sizes, int n_in,
                              void* d_out, int out_size, void* d_ws, size_t ws_size,
                              hipStream_t stream) {
    const float* A = (const float*)d_in[0];   // [8,1024,4096] == [8192,4096]
    const float* W = (const float*)d_in[1];   // [4096,4096] stored [N,K]
    float* C = (float*)d_out;                 // [8192,4096] f32

    const int K = 4096;
    const int N = 4096;
    const int M = in_sizes[0] / K;            // 8192

    const size_t eA = (size_t)M * K;          // 33.5M elements
    const size_t eW = (size_t)N * K;          // 16.8M elements
    const size_t need = 2 * (eA + eW) * 2;    // hi+lo, 2 bytes each = 201 MB

    if (ws_size >= need) {
        char* ws = (char*)d_ws;
        __bf16* Ahp = (__bf16*)(ws);
        __bf16* Alp = (__bf16*)(ws + eA * 2);
        __bf16* Whp = (__bf16*)(ws + eA * 4);
        __bf16* Wlp = (__bf16*)(ws + eA * 4 + eW * 2);

        split_f32_bf16<<<dim3((unsigned)(eA / (256 * 8))), 256, 0, stream>>>(A, Ahp, Alp);
        split_f32_bf16<<<dim3((unsigned)(eW / (256 * 8))), 256, 0, stream>>>(W, Whp, Wlp);

        dim3 grid(N / GBN, M / GBM);          // (16, 64)
        ag_gemm_bf16_async<<<grid, 256, NSTAGES * STAGE_B, stream>>>(
            Ahp, Alp, Whp, Wlp, C, M, N, K);
    } else {
        dim3 grid(N / FBN, M / FBM);
        ag_gemm_f32split_wmma<<<grid, 256, 0, stream>>>(A, W, C, M, N, K);
    }
}
